// GATNet_46084999086804
// MI455X (gfx1250) — matrix operand order repaired and checked by reading, not compiled, not run on hardware
//
#include <hip/hip_runtime.h>
#include <hip/hip_bf16.h>
#include <cstdint>
#include <cstddef>

typedef __attribute__((ext_vector_type(16))) _Float16 v16h;
typedef __attribute__((ext_vector_type(8)))  float    v8f;

// ---------------------------------------------------------------------------
// Utility
// ---------------------------------------------------------------------------
__global__ void zero_f32(float* __restrict__ p, int n) {
    for (int i = blockIdx.x * blockDim.x + threadIdx.x; i < n;
         i += gridDim.x * blockDim.x)
        p[i] = 0.f;
}

// ---------------------------------------------------------------------------
// BatchNorm statistics: per-feature sum / sumsq over M rows of [M,128]
// ---------------------------------------------------------------------------
__global__ void bn_stats(const float* __restrict__ X, int M,
                         float* __restrict__ sums, float* __restrict__ sumsq) {
    int c = threadIdx.x;                       // 0..127, coalesced across row
    float s = 0.f, q = 0.f;
    for (int r = blockIdx.x; r < M; r += gridDim.x) {
        float v = X[(size_t)r * 128 + c];
        s += v; q += v * v;
    }
    atomicAdd(&sums[c], s);
    atomicAdd(&sumsq[c], q);
}

// Fold BN (training-mode batch stats) into the following Linear:
//   BN(x) @ W + bias  ==  x @ (s.W) + (t @ W + bias)
__global__ void bn_fold(const float* __restrict__ sums,
                        const float* __restrict__ sumsq, float invM,
                        const float* __restrict__ g, const float* __restrict__ b,
                        const float* __restrict__ W,
                        const float* __restrict__ bias_in,   // may be null
                        float* __restrict__ Wp, float* __restrict__ bp) {
    __shared__ float ss[128], tt[128];
    int c = threadIdx.x;
    float mu  = sums[c] * invM;
    float var = sumsq[c] * invM - mu * mu;
    float sc  = g[c] * rsqrtf(var + 1e-5f);
    float tc  = b[c] - mu * sc;
    ss[c] = sc; tt[c] = tc;
    __syncthreads();
    float acc = bias_in ? bias_in[c] : 0.f;
    for (int k = 0; k < 128; ++k) {
        float w = W[k * 128 + c];
        Wp[k * 128 + c] = ss[k] * w;
        acc += tt[k] * w;
    }
    bp[c] = acc;
}

// ---------------------------------------------------------------------------
// WMMA GEMM: C[M,128] = A[M,128] @ B[128,128] + bias (optional ReLU)
// A,B f32 in memory, converted to f16 fragments, f32 WMMA accumulate.
// Block = 256 threads = 8 waves; wave w owns rows [blk*128 + w*16, +16).
// A loads are unconditional (row index clamped) 128-bit loads so the
// compiler can batch them; only the stores are range-guarded.
// ---------------------------------------------------------------------------
template <bool RELU>
__global__ __launch_bounds__(256)
void gemm128_wmma(const float* __restrict__ A, const float* __restrict__ B,
                  const float* __restrict__ bias, float* __restrict__ C, int M) {
    __shared__ _Float16 Bt[128 * 132];         // transposed, padded stride
    const int tid   = threadIdx.x;
    const int lane  = tid & 31;
    const int wave  = tid >> 5;

    // Stage B (row-major [k][n] f32) into LDS as f16 [n][k]
    for (int i = tid; i < 128 * 128; i += 256) {
        int k = i >> 7, n = i & 127;
        Bt[n * 132 + k] = (_Float16)B[i];
    }
    __syncthreads();

    const int row0  = blockIdx.x * 128 + wave * 16;
    const int halfq = lane >> 4;
    const int l16   = lane & 15;
    int mrow = row0 + l16;
    if (mrow >= M) mrow = M - 1;               // clamp -> unconditional loads
    const float* __restrict__ arow = A + (size_t)mrow * 128;

    v8f acc[8];
#pragma unroll
    for (int j = 0; j < 8; ++j) acc[j] = (v8f){0.f,0.f,0.f,0.f,0.f,0.f,0.f,0.f};

#pragma unroll
    for (int kk = 0; kk < 128; kk += 32) {
        // A fragment (16-bit A 16x32 layout, ISA 7.12.2):
        // lane needs floats [kk+8h, +8) and [kk+16+8h, +8) -> 4x b128 loads
        const float4 c0 = *(const float4*)(arow + kk + 8 * halfq);
        const float4 c1 = *(const float4*)(arow + kk + 8 * halfq + 4);
        const float4 c2 = *(const float4*)(arow + kk + 16 + 8 * halfq);
        const float4 c3 = *(const float4*)(arow + kk + 16 + 8 * halfq + 4);
        v16h a;
        a[0]  = (_Float16)c0.x; a[1]  = (_Float16)c0.y;
        a[2]  = (_Float16)c0.z; a[3]  = (_Float16)c0.w;
        a[4]  = (_Float16)c1.x; a[5]  = (_Float16)c1.y;
        a[6]  = (_Float16)c1.z; a[7]  = (_Float16)c1.w;
        a[8]  = (_Float16)c2.x; a[9]  = (_Float16)c2.y;
        a[10] = (_Float16)c2.z; a[11] = (_Float16)c2.w;
        a[12] = (_Float16)c3.x; a[13] = (_Float16)c3.y;
        a[14] = (_Float16)c3.z; a[15] = (_Float16)c3.w;

#pragma unroll
        for (int j = 0; j < 8; ++j) {
            // B fragment: same K order as A: [kk+8h, +8) then [kk+16+8h, +8)
            const _Float16* bsrc = &Bt[(j * 16 + l16) * 132 + kk + 8 * halfq];
            v16h bfrag;
#pragma unroll
            for (int v = 0; v < 8; ++v) {
                bfrag[v]     = bsrc[v];
                bfrag[8 + v] = bsrc[16 + v];
            }
            acc[j] = __builtin_amdgcn_wmma_f32_16x16x32_f16(
                false, a, false, bfrag, (short)0, acc[j], false, false);
        }
    }

    // Epilogue: C/D layout — lane half selects M group, VGPR r -> M = 8*half + r
#pragma unroll
    for (int j = 0; j < 8; ++j) {
        int n = j * 16 + l16;
        float bs = bias[n];
#pragma unroll
        for (int r = 0; r < 8; ++r) {
            int m = row0 + 8 * halfq + r;
            if (m < M) {
                float v = acc[j][r] + bs;
                if (RELU) v = fmaxf(v, 0.f);
                C[(size_t)m * 128 + n] = v;
            }
        }
    }
}

// ---------------------------------------------------------------------------
// Attention scores: as_[n,h] = <hW[n,h,:], a_src[h,:]>, same for a_dst
// ---------------------------------------------------------------------------
__global__ void att_scores(const float* __restrict__ hW, int Nn,
                           const float* __restrict__ a_src,
                           const float* __restrict__ a_dst,
                           float* __restrict__ as_, float* __restrict__ ad_) {
    int idx = blockIdx.x * blockDim.x + threadIdx.x;
    if (idx >= Nn * 4) return;
    int n = idx >> 2, hh = idx & 3;
    const float* hp  = hW + (size_t)n * 128 + hh * 32;
    const float* av1 = a_src + hh * 32;
    const float* av2 = a_dst + hh * 32;
    float s1 = 0.f, s2 = 0.f;
#pragma unroll
    for (int c = 0; c < 32; ++c) { float v = hp[c]; s1 += v * av1[c]; s2 += v * av2[c]; }
    as_[idx] = s1;
    ad_[idx] = s2;
}

// float <-> monotone-uint key for atomic max over signed floats
__device__ __forceinline__ unsigned fkey(float f) {
    unsigned u = __float_as_uint(f);
    return (u & 0x80000000u) ? ~u : (u | 0x80000000u);
}
__device__ __forceinline__ float funkey(unsigned k) {
    return (k & 0x80000000u) ? __uint_as_float(k & 0x7FFFFFFFu)
                             : __uint_as_float(~k);
}

// Phase 1: segment max of leaky_relu(as[s]+ad[d]) per (dst, head)
__global__ void edge_max(const int* __restrict__ src, const int* __restrict__ dst,
                         int E_, int Nn,
                         const float* __restrict__ as_, const float* __restrict__ ad_,
                         unsigned* __restrict__ mkey) {
    int idx = blockIdx.x * blockDim.x + threadIdx.x;
    if (idx >= (E_ + Nn) * 4) return;
    int e = idx >> 2, hh = idx & 3;
    int s = (e < E_) ? src[e] : (e - E_);
    int d = (e < E_) ? dst[e] : (e - E_);
    float v = as_[s * 4 + hh] + ad_[d * 4 + hh];
    v = (v > 0.f) ? v : 0.2f * v;              // leaky_relu 0.2
    atomicMax(&mkey[d * 4 + hh], fkey(v));
}

// Phase 2: exp(e - max) per edge, segment-sum denominator
__global__ void edge_exp(const int* __restrict__ src, const int* __restrict__ dst,
                         int E_, int Nn,
                         const float* __restrict__ as_, const float* __restrict__ ad_,
                         const unsigned* __restrict__ mkey,
                         float* __restrict__ expw, float* __restrict__ denom) {
    int idx = blockIdx.x * blockDim.x + threadIdx.x;
    if (idx >= (E_ + Nn) * 4) return;
    int e = idx >> 2, hh = idx & 3;
    int s = (e < E_) ? src[e] : (e - E_);
    int d = (e < E_) ? dst[e] : (e - E_);
    float v = as_[s * 4 + hh] + ad_[d * 4 + hh];
    v = (v > 0.f) ? v : 0.2f * v;
    float m  = funkey(mkey[d * 4 + hh]);
    float ex = __expf(v - m);
    expw[idx] = ex;
    atomicAdd(&denom[d * 4 + hh], ex);
}

// Phase 3: out[d] += alpha * hW[s]; one block per edge, 128 channels
__global__ void edge_scatter(const int* __restrict__ src, const int* __restrict__ dst,
                             int E_, int Nn,
                             const float* __restrict__ hW,
                             const float* __restrict__ expw,
                             const float* __restrict__ denom,
                             float* __restrict__ out) {
    int e  = blockIdx.x;
    int hc = threadIdx.x;
    int hh = hc >> 5;
    int s = (e < E_) ? src[e] : (e - E_);
    int d = (e < E_) ? dst[e] : (e - E_);
    float alpha = expw[e * 4 + hh] / (denom[d * 4 + hh] + 1e-16f);
    atomicAdd(&out[(size_t)d * 128 + hc], hW[(size_t)s * 128 + hc] * alpha);
}

__global__ void bias_relu(const float* __restrict__ aggr,
                          const float* __restrict__ bias,
                          float* __restrict__ h, int total) {
    for (int i = blockIdx.x * blockDim.x + threadIdx.x; i < total;
         i += gridDim.x * blockDim.x) {
        int c = i & 127;
        h[i] = fmaxf(aggr[i] + bias[c], 0.f);
    }
}

// global_add_pool to [32,128]
__global__ void pool_kernel(const float* __restrict__ h,
                            const int* __restrict__ batch,
                            float* __restrict__ pooled, int Nn) {
    for (int i = blockIdx.x * blockDim.x + threadIdx.x; i < Nn * 128;
         i += gridDim.x * blockDim.x) {
        int n = i >> 7, c = i & 127;
        atomicAdd(&pooled[batch[n] * 128 + c], h[i]);
    }
}

// ---------------------------------------------------------------------------
// Tiny head: BN_fc -> Linear+ReLU -> BN_hid -> classifier. One block, 128 thr.
// ---------------------------------------------------------------------------
__global__ __launch_bounds__(128)
void head_kernel(const float* __restrict__ pooled,
                 const float* __restrict__ gfc, const float* __restrict__ bfc,
                 const float* __restrict__ linW, const float* __restrict__ linb,
                 const float* __restrict__ ghid, const float* __restrict__ bhid,
                 const float* __restrict__ clsW, const float* __restrict__ clsb,
                 float* __restrict__ out) {
    __shared__ float P[32][128];
    __shared__ float Q[32][128];
    int c = threadIdx.x;
    float s = 0.f, q = 0.f;
    for (int r = 0; r < 32; ++r) {
        float v = pooled[r * 128 + c];
        P[r][c] = v; s += v; q += v * v;
    }
    float mu  = s * (1.f / 32.f);
    float var = q * (1.f / 32.f) - mu * mu;
    float sc  = gfc[c] * rsqrtf(var + 1e-5f);
    float tc  = bfc[c] - mu * sc;
    for (int r = 0; r < 32; ++r) P[r][c] = P[r][c] * sc + tc;
    __syncthreads();
    for (int r = 0; r < 32; ++r) {
        float a = linb[c];
        for (int k = 0; k < 128; ++k) a += P[r][k] * linW[k * 128 + c];
        Q[r][c] = fmaxf(a, 0.f);
    }
    __syncthreads();
    s = 0.f; q = 0.f;
    for (int r = 0; r < 32; ++r) { float v = Q[r][c]; s += v; q += v * v; }
    mu  = s * (1.f / 32.f);
    var = q * (1.f / 32.f) - mu * mu;
    sc  = ghid[c] * rsqrtf(var + 1e-5f);
    tc  = bhid[c] - mu * sc;
    for (int r = 0; r < 32; ++r) Q[r][c] = Q[r][c] * sc + tc;
    __syncthreads();
    for (int idx = c; idx < 32 * 10; idx += 128) {
        int r = idx / 10, cl = idx % 10;
        float a = clsb[cl];
        for (int k = 0; k < 128; ++k) a += Q[r][k] * clsW[k * 10 + cl];
        out[idx] = a;
    }
}

// ---------------------------------------------------------------------------
// Host orchestration
// ---------------------------------------------------------------------------
extern "C" void kernel_launch(void* const* d_in, const int* in_sizes, int n_in,
                              void* d_out, int out_size, void* d_ws, size_t ws_size,
                              hipStream_t stream) {
    const float* x       = (const float*)d_in[0];
    const int*   eidx    = (const int*)  d_in[1];
    const int*   batch   = (const int*)  d_in[2];
    const float* bnf_g   = (const float*)d_in[3];
    const float* bnf_b   = (const float*)d_in[4];
    const float* W_feat  = (const float*)d_in[5];
    const float* b_feat  = (const float*)d_in[6];
    const float* conv_W  = (const float*)d_in[7];
    const float* att_src = (const float*)d_in[8];
    const float* att_dst = (const float*)d_in[9];
    const float* conv_b  = (const float*)d_in[10];
    const float* bnc_g   = (const float*)d_in[11];
    const float* bnc_b   = (const float*)d_in[12];
    const float* lin_W   = (const float*)d_in[13];
    const float* lin_b   = (const float*)d_in[14];
    const float* bnfc_g  = (const float*)d_in[15];
    const float* bnfc_b  = (const float*)d_in[16];
    const float* bnh_g   = (const float*)d_in[17];
    const float* bnh_b   = (const float*)d_in[18];
    const float* cls_W   = (const float*)d_in[19];
    const float* cls_b   = (const float*)d_in[20];

    const int Nn = in_sizes[0] / 128;
    const int E  = in_sizes[1] / 2;
    const int* srcp = eidx;
    const int* dstp = eidx + E;

    // workspace bump allocator (256B aligned slots)
    char*  base = (char*)d_ws;
    size_t off  = 0;
    auto alloc = [&](size_t bytes) -> void* {
        void* p = base + off;
        off = (off + bytes + 255) & ~(size_t)255;
        return p;
    };
    float*    h      = (float*)   alloc((size_t)Nn * 128 * 4);
    float*    hW     = (float*)   alloc((size_t)Nn * 128 * 4);
    float*    aggr   = (float*)   alloc((size_t)Nn * 128 * 4);
    float*    Wp     = (float*)   alloc(128 * 128 * 4);
    float*    bp     = (float*)   alloc(128 * 4);
    float*    sums   = (float*)   alloc(128 * 4);
    float*    sumsq  = (float*)   alloc(128 * 4);
    float*    as_    = (float*)   alloc((size_t)Nn * 4 * 4);
    float*    ad_    = (float*)   alloc((size_t)Nn * 4 * 4);
    unsigned* mkey   = (unsigned*)alloc((size_t)Nn * 4 * 4);
    float*    denom  = (float*)   alloc((size_t)Nn * 4 * 4);
    float*    expw   = (float*)   alloc((size_t)(E + Nn) * 4 * 4);
    float*    pooled = (float*)   alloc(32 * 128 * 4);
    (void)n_in; (void)out_size; (void)ws_size;

    const int  gemmBlocks = (Nn + 127) / 128;
    const int  nh4        = Nn * 4;
    const int  edges4     = (E + Nn) * 4;
    const float invN      = 1.f / (float)Nn;
    const dim3 z1(1024), z2(256);

    // --- input BN folded into feature Linear, then WMMA GEMM + ReLU ---
    zero_f32<<<1, 256, 0, stream>>>(sums, 128);
    zero_f32<<<1, 256, 0, stream>>>(sumsq, 128);
    bn_stats<<<512, 128, 0, stream>>>(x, Nn, sums, sumsq);
    bn_fold<<<1, 128, 0, stream>>>(sums, sumsq, invN, bnf_g, bnf_b,
                                   W_feat, b_feat, Wp, bp);
    gemm128_wmma<true><<<gemmBlocks, 256, 0, stream>>>(x, Wp, bp, h, Nn);

    // --- 3 GAT layers ---
    for (int i = 0; i < 3; ++i) {
        zero_f32<<<1, 256, 0, stream>>>(sums, 128);
        zero_f32<<<1, 256, 0, stream>>>(sumsq, 128);
        bn_stats<<<512, 128, 0, stream>>>(h, Nn, sums, sumsq);
        bn_fold<<<1, 128, 0, stream>>>(sums, sumsq, invN,
                                       bnc_g + i * 128, bnc_b + i * 128,
                                       conv_W + (size_t)i * 128 * 128, nullptr,
                                       Wp, bp);
        gemm128_wmma<false><<<gemmBlocks, 256, 0, stream>>>(h, Wp, bp, hW, Nn);

        att_scores<<<(nh4 + 255) / 256, 256, 0, stream>>>(
            hW, Nn, att_src + i * 128, att_dst + i * 128, as_, ad_);

        zero_f32<<<z1, z2, 0, stream>>>((float*)mkey, nh4);   // key 0 == -inf
        zero_f32<<<z1, z2, 0, stream>>>(denom, nh4);
        zero_f32<<<z1, z2, 0, stream>>>(aggr, Nn * 128);

        edge_max<<<(edges4 + 255) / 256, 256, 0, stream>>>(
            srcp, dstp, E, Nn, as_, ad_, mkey);
        edge_exp<<<(edges4 + 255) / 256, 256, 0, stream>>>(
            srcp, dstp, E, Nn, as_, ad_, mkey, expw, denom);
        edge_scatter<<<E + Nn, 128, 0, stream>>>(
            srcp, dstp, E, Nn, hW, expw, denom, aggr);

        bias_relu<<<z1, z2, 0, stream>>>(aggr, conv_b + i * 128, h, Nn * 128);
    }

    // --- pooling + MLP head ---
    zero_f32<<<1, 256, 0, stream>>>(pooled, 32 * 128);
    pool_kernel<<<z1, z2, 0, stream>>>(h, batch, pooled, Nn);
    head_kernel<<<1, 128, 0, stream>>>(pooled, bnfc_g, bnfc_b, lin_W, lin_b,
                                       bnh_g, bnh_b, cls_W, cls_b,
                                       (float*)d_out);
}
